// FNO1d_FFT_51376398795170
// MI455X (gfx1250) — matrix operand order repaired and checked by reading, not compiled, not run on hardware
//
#include <hip/hip_runtime.h>
#include <hip/hip_bf16.h>

#define B_ 64
#define N_ 8192
#define W_ 64
#define M_ 16
#define L_ 4

typedef __attribute__((ext_vector_type(16))) __bf16          v16bf;
typedef __attribute__((ext_vector_type(8)))  float           v8f;
typedef __attribute__((ext_vector_type(16))) unsigned short  v16us;
typedef __attribute__((ext_vector_type(8)))  unsigned short  v8us;

#if defined(__gfx1250__) && __has_builtin(__builtin_amdgcn_global_load_async_to_lds_b128) && __has_builtin(__builtin_amdgcn_s_wait_asynccnt)
#define USE_ASYNC_LDS 1
#else
#define USE_ASYNC_LDS 0
#endif

// 16-byte global -> LDS copy. On gfx1250 toolchains that expose the async
// builtins this lowers to GLOBAL_LOAD_ASYNC_TO_LDS_B128 (ASYNCcnt-tracked,
// no VGPR staging); otherwise falls back to b128 load + ds_store_b128.
#if USE_ASYNC_LDS
typedef int async_v4i __attribute__((vector_size(16)));
typedef __attribute__((address_space(1))) async_v4i* g_b128_ptr;
typedef __attribute__((address_space(3))) async_v4i* l_b128_ptr;
#endif

static __device__ __forceinline__ void cp16_g2l(unsigned short* lds_dst,
                                                const unsigned short* gsrc) {
#if USE_ASYNC_LDS
  __builtin_amdgcn_global_load_async_to_lds_b128(
      (g_b128_ptr)gsrc, (l_b128_ptr)lds_dst, 0, 0);
#else
  *(v8us*)lds_dst = *(const v8us*)gsrc;
#endif
}

static __device__ __forceinline__ void cp_fence() {
#if USE_ASYNC_LDS
  __builtin_amdgcn_s_wait_asynccnt(0);
#endif
}

static __device__ __forceinline__ unsigned short f2bf(float f) {
  unsigned int u = __builtin_bit_cast(unsigned int, f);
  u += 0x7FFFu + ((u >> 16) & 1u);            // round-to-nearest-even
  return (unsigned short)(u >> 16);
}

static __device__ __forceinline__ float gelu_exact(float x) {
  return 0.5f * x * (1.0f + erff(x * 0.70710678118654752f));
}

static __device__ __forceinline__ v8f wmma_bf16(v16us a, v16us b, v8f c) {
  return __builtin_amdgcn_wmma_f32_16x16x32_bf16(
      false, __builtin_bit_cast(v16bf, a),
      false, __builtin_bit_cast(v16bf, b),
      (short)0, c, false, false);
}

#define SHUF16(lo, hi) __builtin_shufflevector(lo, hi, 0,1,2,3,4,5,6,7,8,9,10,11,12,13,14,15)

// ---------------------------------------------------------------------------
// K0: build bf16 DFT tables + bf16 weight copies.
//   T1t[c][n]: c<16 -> cos(2*pi*c*n/N); c>=16 -> -sin(2*pi*(c-16)*n/N)   (rfft)
//   T2 [c][n]: c<16 -> (c==0?1:2cos)/N ; c>=16 -> (k==0?0:-2sin)/N       (irfft)
// ---------------------------------------------------------------------------
__global__ void __launch_bounds__(256) k_init(const float* __restrict__ cw,
                                              const float* __restrict__ fc1w,
                                              unsigned short* __restrict__ t1t,
                                              unsigned short* __restrict__ t2,
                                              unsigned short* __restrict__ cwbf,
                                              unsigned short* __restrict__ w1bf) {
  int idx = blockIdx.x * 256 + threadIdx.x;     // 0 .. 32*N-1
  int c = idx >> 13, n = idx & (N_ - 1);
  int k = c & 15;
  int p = (k * n) & (N_ - 1);                   // exact (power-of-two N)
  float th = 6.28318530717958647f * (float)p * (1.0f / (float)N_);
  float sn, cs;
  __sincosf(th, &sn, &cs);
  t1t[idx] = f2bf(c < 16 ? cs : -sn);
  float v2;
  if (c < 16) v2 = (c == 0 ? 1.0f : 2.0f * cs) * (1.0f / (float)N_);
  else        v2 = (k == 0 ? 0.0f : -2.0f * sn) * (1.0f / (float)N_);
  t2[idx] = f2bf(v2);
  if (idx < L_ * W_ * W_) cwbf[idx] = f2bf(cw[idx]);
  if (idx < W_ * 128)     w1bf[idx] = f2bf(fc1w[idx]);
}

// ---------------------------------------------------------------------------
// K1: fc0 lift:  h[b][i][n] = x[b][n]*w[0][i] + grid(n)*w[1][i] + bias[i]
// ---------------------------------------------------------------------------
__global__ void __launch_bounds__(256) k_fc0(const float* __restrict__ x,
                                             const float* __restrict__ w,
                                             const float* __restrict__ bias,
                                             unsigned short* __restrict__ h) {
  int tid = blockIdx.x * 256 + threadIdx.x;     // B*N threads
  int b = tid >> 13, n = tid & (N_ - 1);
  float xv = x[b * N_ + n];
  float g  = (float)n * (1.0f / (float)(N_ - 1));
  unsigned short* hp = h + (size_t)b * W_ * N_ + n;
#pragma unroll 8
  for (int i = 0; i < W_; ++i) {
    hp[(size_t)i * N_] = f2bf(xv * w[i] + g * w[W_ + i] + bias[i]);
  }
}

// ---------------------------------------------------------------------------
// K2: forward truncated DFT:  HF[r][c] = sum_n h[r][n] * T1t[c][n]
//     rows r = b*64+i (tiles of 16), cols c in [0,32), K = N split over 4 waves.
// ---------------------------------------------------------------------------
__global__ void __launch_bounds__(256) k_fwd_dft(const unsigned short* __restrict__ h,
                                                 const unsigned short* __restrict__ t1t,
                                                 float* __restrict__ hf) {
  __shared__ float acc[2][16][16];
  int t = threadIdx.x, lane = t & 31, wv = t >> 5;
  int ct = wv & 1, ks = wv >> 1;                // ct: col tile, ks: K slice (0..3)
  int col = lane & 15, hi = lane >> 4;
  int rowbase = blockIdx.x * 16;

  ((float*)acc)[t] = 0.0f;
  ((float*)acc)[t + 256] = 0.0f;
  __syncthreads();

  const unsigned short* arow = h + (size_t)(rowbase + col) * N_;          // A row = lane&15
  const unsigned short* bcol = t1t + (size_t)(ct * 16 + col) * N_;        // B col = lane&15
  v8f c = {};
  int k0 = ks * (N_ / 4);
#pragma unroll 4
  for (int it = 0; it < (N_ / 4) / 32; ++it, k0 += 32) {
    __builtin_prefetch(arow + k0 + 512, 0, 1);
    v8us a0 = *(const v8us*)(arow + k0 + hi * 8);
    v8us a1 = *(const v8us*)(arow + k0 + 16 + hi * 8);
    v8us b0 = *(const v8us*)(bcol + k0 + hi * 8);
    v8us b1 = *(const v8us*)(bcol + k0 + 16 + hi * 8);
    c = wmma_bf16(SHUF16(a0, a1), SHUF16(b0, b1), c);
  }
#pragma unroll
  for (int j = 0; j < 8; ++j)
    atomicAdd(&acc[ct][j + 8 * hi][col], c[j]);     // ds_add_f32 reduction
  __syncthreads();
#pragma unroll
  for (int r = 0; r < 2; ++r) {
    int idx = r * 256 + t;
    int m = idx >> 5, cc = idx & 31;
    hf[(size_t)(rowbase + m) * 32 + cc] = acc[cc >> 4][m][cc & 15];
  }
}

// ---------------------------------------------------------------------------
// K2b: per-mode complex 64x64 channel mix (tiny: 33 MFLOP total) -> bf16 OF.
//   OF[b][o][k]    = sum_i hr*wr - hi*wi
//   OF[b][o][16+k] = sum_i hr*wi + hi*wr
// ---------------------------------------------------------------------------
__global__ void __launch_bounds__(256) k_modemix(const float* __restrict__ hf,
                                                 const float* __restrict__ swr,
                                                 const float* __restrict__ swi,
                                                 int l,
                                                 unsigned short* __restrict__ ofbf) {
  int b = blockIdx.x, kk = blockIdx.y;
  int t = threadIdx.x;
  int o = t & 63, k = kk * 4 + (t >> 6);
  const float* hfb = hf + (size_t)b * W_ * 32;
  const float* wr = swr + (size_t)l * W_ * W_ * M_ + k;
  const float* wi = swi + (size_t)l * W_ * W_ * M_ + k;
  float orr = 0.0f, oii = 0.0f;
#pragma unroll 4
  for (int i = 0; i < W_; ++i) {
    float hr = hfb[i * 32 + k];
    float hq = hfb[i * 32 + 16 + k];
    float a = wr[(i * 64 + o) * 16];
    float d = wi[(i * 64 + o) * 16];
    orr += hr * a - hq * d;
    oii += hr * d + hq * a;
  }
  ofbf[((size_t)b * W_ + o) * 32 + k]      = f2bf(orr);
  ofbf[((size_t)b * W_ + o) * 32 + 16 + k] = f2bf(oii);
}

// ---------------------------------------------------------------------------
// K3: fused {inverse DFT + 1x1 conv + bias + GELU}, in-place on h.
//   Block = (batch b, 128-column chunk). 8 waves, one 16-col tile each.
//   out[o][n] = sum_c OF[o][c]*T2[c][n] + sum_i CW[o][i]*h[i][n] + cb[o]
// ---------------------------------------------------------------------------
__global__ void __launch_bounds__(256) k_spectral(unsigned short* __restrict__ h,
                                                  const unsigned short* __restrict__ ofbf,
                                                  const unsigned short* __restrict__ t2,
                                                  const unsigned short* __restrict__ cwl,
                                                  const float* __restrict__ cbl,
                                                  int do_gelu) {
  __shared__ __align__(16) unsigned short sh_h[64][128];
  __shared__ __align__(16) unsigned short sh_t2[32][128];
  __shared__ __align__(16) unsigned short sh_cw[64][64];
  __shared__ __align__(16) unsigned short sh_of[64][32];
  __shared__ float sh_cb[64];
  int t = threadIdx.x;
  int b = blockIdx.y, n0 = blockIdx.x * 128;
  unsigned short* hb = h + (size_t)b * W_ * N_;

#pragma unroll
  for (int r = 0; r < 4; ++r) {                 // h tile: 64x128 bf16
    int idx = (r * 256 + t) * 8;
    int row = idx >> 7, cp = idx & 127;
    cp16_g2l(&sh_h[row][cp], hb + (size_t)row * N_ + n0 + cp);
  }
#pragma unroll
  for (int r = 0; r < 2; ++r) {                 // T2 tile: 32x128
    int idx = (r * 256 + t) * 8;
    int row = idx >> 7, cp = idx & 127;
    cp16_g2l(&sh_t2[row][cp], t2 + (size_t)row * N_ + n0 + cp);
  }
#pragma unroll
  for (int r = 0; r < 2; ++r) {                 // conv weights 64x64
    int idx = (r * 256 + t) * 8;
    cp16_g2l(&sh_cw[0][0] + idx, cwl + idx);
  }
  {                                             // OF[b]: 64x32
    int idx = t * 8;
    cp16_g2l(&sh_of[0][0] + idx, ofbf + (size_t)b * W_ * 32 + idx);
  }
  if (t < 64) sh_cb[t] = cbl[t];                // bias
  cp_fence();                                   // s_wait_asynccnt 0
  __syncthreads();

  int lane = t & 31, ct = t >> 5;
  int col = lane & 15, hi = lane >> 4;

  // Hoisted B fragments (independent of output row tile).
  v16us bh0, bh1, bt2;
#pragma unroll
  for (int e = 0; e < 16; ++e) {
    bh0[e] = sh_h[hi * 8 + (e & 7) + (e >> 3) * 16][ct * 16 + col];
    bh1[e] = sh_h[32 + hi * 8 + (e & 7) + (e >> 3) * 16][ct * 16 + col];
    bt2[e] = sh_t2[hi * 8 + (e & 7) + (e >> 3) * 16][ct * 16 + col];
  }

#pragma unroll
  for (int ot = 0; ot < 4; ++ot) {
    int ar = ot * 16 + col;                     // A row = lane&15
    v8us f0, f1;
    f0 = *(const v8us*)&sh_of[ar][hi * 8];
    f1 = *(const v8us*)&sh_of[ar][16 + hi * 8];
    v16us aof = SHUF16(f0, f1);
    f0 = *(const v8us*)&sh_cw[ar][hi * 8];
    f1 = *(const v8us*)&sh_cw[ar][16 + hi * 8];
    v16us ac0 = SHUF16(f0, f1);
    f0 = *(const v8us*)&sh_cw[ar][32 + hi * 8];
    f1 = *(const v8us*)&sh_cw[ar][48 + hi * 8];
    v16us ac1 = SHUF16(f0, f1);

    v8f acc = {};
    acc = wmma_bf16(aof, bt2, acc);             // inverse DFT (K=32)
    acc = wmma_bf16(ac0, bh0, acc);             // 1x1 conv, K 0..31
    acc = wmma_bf16(ac1, bh1, acc);             // 1x1 conv, K 32..63

#pragma unroll
    for (int j = 0; j < 8; ++j) {
      int o = ot * 16 + j + 8 * hi;
      float v = acc[j] + sh_cb[o];
      if (do_gelu) v = gelu_exact(v);
      hb[((size_t)o) * N_ + n0 + ct * 16 + col] = f2bf(v);
    }
  }
}

// ---------------------------------------------------------------------------
// K4: head: out[b][n] = fc2( gelu( h[b][:,n] @ fc1 + b1 ) ) + b2
// ---------------------------------------------------------------------------
__global__ void __launch_bounds__(256) k_head(const unsigned short* __restrict__ h,
                                              const unsigned short* __restrict__ w1bf,
                                              const float* __restrict__ b1,
                                              const float* __restrict__ w2,
                                              const float* __restrict__ b2,
                                              float* __restrict__ out) {
  __shared__ __align__(16) unsigned short sh_h[64][64];
  __shared__ __align__(16) unsigned short sh_w1[64][128];
  __shared__ float sh_out[64];
  int t = threadIdx.x;
  int b = blockIdx.y, n0 = blockIdx.x * 64;
  const unsigned short* hb = h + (size_t)b * W_ * N_;

#pragma unroll
  for (int r = 0; r < 2; ++r) {                 // h tile 64x64
    int idx = (r * 256 + t) * 8;
    int row = idx >> 6, cp = idx & 63;
    cp16_g2l(&sh_h[row][cp], hb + (size_t)row * N_ + n0 + cp);
  }
#pragma unroll
  for (int r = 0; r < 4; ++r) {                 // fc1 weights 64x128
    int idx = (r * 256 + t) * 8;
    cp16_g2l(&sh_w1[0][0] + idx, w1bf + idx);
  }
  if (t < 64) sh_out[t] = 0.0f;
  cp_fence();                                   // s_wait_asynccnt 0
  __syncthreads();

  int lane = t & 31, ct = t >> 5;
  int col = lane & 15, hi = lane >> 4;

  v16us bw0, bw1;
#pragma unroll
  for (int e = 0; e < 16; ++e) {
    bw0[e] = sh_w1[hi * 8 + (e & 7) + (e >> 3) * 16][ct * 16 + col];
    bw1[e] = sh_w1[32 + hi * 8 + (e & 7) + (e >> 3) * 16][ct * 16 + col];
  }
  float bias1 = b1[ct * 16 + col];
  float f2v   = w2[ct * 16 + col];

#pragma unroll
  for (int rt = 0; rt < 4; ++rt) {
    v16us a0, a1;                               // A rows = n positions (LDS transpose gather)
#pragma unroll
    for (int e = 0; e < 8; ++e) {
      a0[e]     = sh_h[hi * 8 + e][rt * 16 + col];
      a0[e + 8] = sh_h[16 + hi * 8 + e][rt * 16 + col];
      a1[e]     = sh_h[32 + hi * 8 + e][rt * 16 + col];
      a1[e + 8] = sh_h[48 + hi * 8 + e][rt * 16 + col];
    }
    v8f acc = {};
    acc = wmma_bf16(a0, bw0, acc);
    acc = wmma_bf16(a1, bw1, acc);
#pragma unroll
    for (int j = 0; j < 8; ++j) {
      float g = gelu_exact(acc[j] + bias1);
      atomicAdd(&sh_out[rt * 16 + j + 8 * hi], g * f2v);   // fc2 matvec reduce
    }
  }
  __syncthreads();
  if (t < 64) out[(size_t)b * N_ + n0 + t] = sh_out[t] + b2[0];
}

// ---------------------------------------------------------------------------
extern "C" void kernel_launch(void* const* d_in, const int* in_sizes, int n_in,
                              void* d_out, int out_size, void* d_ws, size_t ws_size,
                              hipStream_t stream) {
  (void)in_sizes; (void)n_in; (void)out_size; (void)ws_size;
  const float* x    = (const float*)d_in[0];
  const float* sw_r = (const float*)d_in[1];
  const float* sw_i = (const float*)d_in[2];
  const float* cw   = (const float*)d_in[3];
  const float* cb   = (const float*)d_in[4];
  const float* fc0w = (const float*)d_in[5];
  const float* fc0b = (const float*)d_in[6];
  const float* fc1w = (const float*)d_in[7];
  const float* fc1b = (const float*)d_in[8];
  const float* fc2w = (const float*)d_in[9];
  const float* fc2b = (const float*)d_in[10];
  float* out = (float*)d_out;

  char* ws = (char*)d_ws;
  size_t off = 0;
  auto alloc = [&](size_t bytes) {
    char* p = ws + off;
    off += (bytes + 255) & ~(size_t)255;
    return p;
  };
  unsigned short* hbuf = (unsigned short*)alloc((size_t)B_ * W_ * N_ * 2);  // 64 MB
  float*          HF   = (float*)alloc((size_t)B_ * W_ * 32 * 4);
  unsigned short* OF   = (unsigned short*)alloc((size_t)B_ * W_ * 32 * 2);
  unsigned short* T1   = (unsigned short*)alloc((size_t)32 * N_ * 2);
  unsigned short* T2   = (unsigned short*)alloc((size_t)32 * N_ * 2);
  unsigned short* CWB  = (unsigned short*)alloc((size_t)L_ * W_ * W_ * 2);
  unsigned short* W1B  = (unsigned short*)alloc((size_t)W_ * 128 * 2);

  k_init<<<(32 * N_) / 256, 256, 0, stream>>>(cw, fc1w, T1, T2, CWB, W1B);
  k_fc0<<<(B_ * N_) / 256, 256, 0, stream>>>(x, fc0w, fc0b, hbuf);

  for (int l = 0; l < L_; ++l) {
    k_fwd_dft<<<(B_ * W_) / 16, 256, 0, stream>>>(hbuf, T1, HF);
    k_modemix<<<dim3(B_, 4), 256, 0, stream>>>(HF, sw_r, sw_i, l, OF);
    k_spectral<<<dim3(N_ / 128, B_), 256, 0, stream>>>(
        hbuf, OF, T2, CWB + (size_t)l * W_ * W_, cb + l * W_, (l < L_ - 1) ? 1 : 0);
  }
  k_head<<<dim3(N_ / 64, B_), 256, 0, stream>>>(hbuf, W1B, fc1b, fc2w, fc2b, out);
}
